// VectorQuantizer_57337813401660
// MI455X (gfx1250) — matrix operand order repaired and checked by reading, not compiled, not run on hardware
//
#include <hip/hip_runtime.h>

// ---------------------------------------------------------------------------
// VQ-VAE quantizer for MI455X (gfx1250, wave32).
// Distance GEMM on the bf16 matrix pipe (V_WMMA_F32_16X16X32_BF16) using a
// split hi/lo bf16 decomposition of the fp32 inputs:
//   z.e = zh.eh + zh.el + zl.eh + zl.el   (f32 accumulate)
// Residual error ~2^-17 relative (below fp32 summation-order noise) -> argmin
// preserved. 17.2 GFLOP compute vs ~65MB HBM traffic -> compute bound.
// Each wave processes 2 code-tiles per pass, reusing A fragments: per 32-chan
// chunk = 4 ds_load_b128 + 8 global_load_b128 feeding 8 WMMAs (1.5 loads/WMMA
// vs 2.0 before), easing LDS/L2 pressure against the XDL pipe.
// ---------------------------------------------------------------------------

typedef __attribute__((ext_vector_type(16))) __bf16      v16bf;
typedef __attribute__((ext_vector_type(8)))  float       v8f;
typedef __attribute__((ext_vector_type(4)))  unsigned int u4;

#define NUM_EMB 1024
#define EMB_DIM 256
#define CPAIRS  128                  // EMB_DIM/2 packed bf16 dwords per row
#define HW      1024                 // 32*32 positions per image
#define ROWS_PER_BLOCK 128           // 8 waves * 16 rows
#define LDS_STRIDE 132               // 128 + 4 dword pad (conflict-free b128)

union Frag { u4 q[2]; v16bf v; };

__device__ inline unsigned short f32_to_bf16_rne(float f) {
  unsigned int x = __float_as_uint(f);
  unsigned int r = x + 0x7FFFu + ((x >> 16) & 1u);   // round-to-nearest-even
  return (unsigned short)(r >> 16);
}
__device__ inline float bf16_to_f32(unsigned short h) {
  return __uint_as_float(((unsigned int)h) << 16);
}
__device__ inline void split_pack(float f0, float f1,
                                  unsigned int* hi, unsigned int* lo) {
  unsigned short h0 = f32_to_bf16_rne(f0);
  unsigned short h1 = f32_to_bf16_rne(f1);
  unsigned short l0 = f32_to_bf16_rne(f0 - bf16_to_f32(h0));
  unsigned short l1 = f32_to_bf16_rne(f1 - bf16_to_f32(h1));
  *hi = (unsigned int)h0 | ((unsigned int)h1 << 16);
  *lo = (unsigned int)l0 | ((unsigned int)l1 << 16);
}

#define WMMA_BF16(A, B, C) \
  __builtin_amdgcn_wmma_f32_16x16x32_bf16(false, (A), false, (B), (short)0, (C), false, false)

// ------- Kernel A: pack emb into hi/lo bf16 planes, ||e||^2, zero loss -----
__global__ void __launch_bounds__(128)
vq_prep_kernel(const float* __restrict__ emb,
               unsigned int* __restrict__ embHi,
               unsigned int* __restrict__ embLo,
               float* __restrict__ enorm,
               double* __restrict__ lossAcc) {
  const int code = blockIdx.x;
  const int t = threadIdx.x;                 // channel pair 0..127
  if (code == 0 && t == 0) *lossAcc = 0.0;
  float f0 = emb[code * EMB_DIM + 2 * t];
  float f1 = emb[code * EMB_DIM + 2 * t + 1];
  unsigned int hi, lo;
  split_pack(f0, f1, &hi, &lo);
  embHi[code * CPAIRS + t] = hi;
  embLo[code * CPAIRS + t] = lo;
  float s = f0 * f0 + f1 * f1;               // ||e||^2 from original fp32
#pragma unroll
  for (int m = 16; m >= 1; m >>= 1) s += __shfl_xor(s, m, 32);
  __shared__ float ws[4];
  if ((t & 31) == 0) ws[t >> 5] = s;
  __syncthreads();
  if (t == 0) enorm[code] = ws[0] + ws[1] + ws[2] + ws[3];
}

// ------- Kernel B: split-bf16 WMMA distance matmul + per-row argmin --------
// 256 threads = 8 waves, 128 z-rows; each wave owns 16 rows and scans the
// 1024 codes 32 at a time (2 N-tiles sharing the A fragments).
__global__ void __launch_bounds__(256)
vq_argmin_kernel(const float* __restrict__ z,
                 const unsigned int* __restrict__ embHi,
                 const unsigned int* __restrict__ embLo,
                 const float* __restrict__ enorm,
                 int* __restrict__ indices) {
  extern __shared__ unsigned int lds[];                 // hi + lo tiles
  unsigned int* tileHi = lds;                           // [128][LDS_STRIDE]
  unsigned int* tileLo = lds + ROWS_PER_BLOCK * LDS_STRIDE;

  const int t    = threadIdx.x;
  const int lane = t & 31;
  const int wave = t >> 5;
  const int b    = blockIdx.x >> 3;                     // 8 blocks per image
  const int p0   = (blockIdx.x & 7) * ROWS_PER_BLOCK;
  const float* zimg = z + (size_t)b * (EMB_DIM * HW);

  // Stage tile, converting fp32 -> packed hi/lo bf16:
  //   tileHi[m][cp] packs channels (2cp, 2cp+1) of row p0+m.
  {
    const int m = t & 127;
    const int cpHalf = t >> 7;                          // 0 or 1
#pragma unroll 4
    for (int it = 0; it < CPAIRS / 2; ++it) {
      int cp = it * 2 + cpHalf;
      int c = cp * 2;
      float f0 = zimg[c * HW + p0 + m];                 // coalesced per chan
      float f1 = zimg[(c + 1) * HW + p0 + m];
      unsigned int hi, lo;
      split_pack(f0, f1, &hi, &lo);
      tileHi[m * LDS_STRIDE + cp] = hi;
      tileLo[m * LDS_STRIDE + cp] = lo;
    }
  }
  __syncthreads();

  // 16-bit A 16x32 frag (8 VGPRs): lane L -> row M=L%16; sel=L/16 picks the
  // K-half: dwords ch*16 + sel*8 + {0..7}.  B 32x16 frag: lane L ->
  // col N=L%16, K = ch*32 + sel*16 + 0..15 -> 8 contiguous packed dwords.
  const int mrow = wave * 16 + (lane & 15);
  const int sel  = lane >> 4;
  const unsigned int* aHiRow = tileHi + mrow * LDS_STRIDE;
  const unsigned int* aLoRow = tileLo + mrow * LDS_STRIDE;
  const int codeLane = lane & 15;

  float minv[8];
  int   mini[8];
#pragma unroll
  for (int i = 0; i < 8; ++i) { minv[i] = 3.4e38f; mini[i] = 0; }

  for (int nt = 0; nt < NUM_EMB / 32; ++nt) {
    const int code0 = nt * 32 + codeLane;
    const int code1 = code0 + 16;
    const float en0 = enorm[code0];
    const float en1 = enorm[code1];
    const unsigned int* bHi0 = embHi + (size_t)code0 * CPAIRS + sel * 8;
    const unsigned int* bLo0 = embLo + (size_t)code0 * CPAIRS + sel * 8;
    const unsigned int* bHi1 = embHi + (size_t)code1 * CPAIRS + sel * 8;
    const unsigned int* bLo1 = embLo + (size_t)code1 * CPAIRS + sel * 8;
    v8f acc0 = {0.f, 0.f, 0.f, 0.f, 0.f, 0.f, 0.f, 0.f};
    v8f acc1 = acc0;
#pragma unroll 2
    for (int ch = 0; ch < EMB_DIM / 32; ++ch) {
      const int ab = ch * 16 + sel * 8;
      Frag ahi, alo, bhi0, blo0, bhi1, blo1;
      ahi.q[0]  = *(const u4*)(aHiRow + ab);            // ds_load_b128 x4
      ahi.q[1]  = *(const u4*)(aHiRow + ab + 4);
      alo.q[0]  = *(const u4*)(aLoRow + ab);
      alo.q[1]  = *(const u4*)(aLoRow + ab + 4);
      bhi0.q[0] = *(const u4*)(bHi0 + ch * 16);         // global_load_b128 x8
      bhi0.q[1] = *(const u4*)(bHi0 + ch * 16 + 4);
      blo0.q[0] = *(const u4*)(bLo0 + ch * 16);
      blo0.q[1] = *(const u4*)(bLo0 + ch * 16 + 4);
      bhi1.q[0] = *(const u4*)(bHi1 + ch * 16);
      bhi1.q[1] = *(const u4*)(bHi1 + ch * 16 + 4);
      blo1.q[0] = *(const u4*)(bLo1 + ch * 16);
      blo1.q[1] = *(const u4*)(bLo1 + ch * 16 + 4);
      // A fragments reused across both code tiles: 8 WMMAs / 12 loads
      acc0 = WMMA_BF16(ahi.v, bhi0.v, acc0);
      acc0 = WMMA_BF16(ahi.v, blo0.v, acc0);
      acc0 = WMMA_BF16(alo.v, bhi0.v, acc0);
      acc0 = WMMA_BF16(alo.v, blo0.v, acc0);
      acc1 = WMMA_BF16(ahi.v, bhi1.v, acc1);
      acc1 = WMMA_BF16(ahi.v, blo1.v, acc1);
      acc1 = WMMA_BF16(alo.v, bhi1.v, acc1);
      acc1 = WMMA_BF16(alo.v, blo1.v, acc1);
    }
    // D layout: VGPR i -> row (wave*16 + i + 8*sel), col = code0 / code1.
    // code0 < code1: compare in ascending order for first-min tiebreak.
#pragma unroll
    for (int i = 0; i < 8; ++i) {
      float d0 = en0 - 2.0f * acc0[i];       // ||e||^2 - 2 z.e (||z||^2 const)
      if (d0 < minv[i]) { minv[i] = d0; mini[i] = code0; }
      float d1 = en1 - 2.0f * acc1[i];
      if (d1 < minv[i]) { minv[i] = d1; mini[i] = code1; }
    }
  }

  // Argmin across the 16 lanes of each half-wave (cols), first-min tiebreak.
#pragma unroll
  for (int m = 1; m < 16; m <<= 1) {
#pragma unroll
    for (int i = 0; i < 8; ++i) {
      float ov = __shfl_xor(minv[i], m, 32);
      int   oi = __shfl_xor(mini[i], m, 32);
      if (ov < minv[i] || (ov == minv[i] && oi < mini[i])) {
        minv[i] = ov; mini[i] = oi;
      }
    }
  }

  const int rowBase = blockIdx.x * ROWS_PER_BLOCK + wave * 16;
  if (lane == 0) {
#pragma unroll
    for (int i = 0; i < 8; ++i) indices[rowBase + i] = mini[i];
  } else if (lane == 16) {
#pragma unroll
    for (int i = 0; i < 8; ++i) indices[rowBase + 8 + i] = mini[i];
  }
}

// ------- Kernel C: gather codebook rows (orig fp32), write out, loss sum ---
__global__ void vq_gather_kernel(const float* __restrict__ z,
                                 const float* __restrict__ emb,
                                 const int* __restrict__ indices,
                                 float* __restrict__ out,
                                 double* __restrict__ lossAcc,
                                 int nElem) {
  int base = (blockIdx.x * blockDim.x + threadIdx.x) * 4;
  float s = 0.f;
  if (base < nElem) {
#pragma unroll
    for (int j = 0; j < 4; ++j) {
      int o = base + j;                       // o = b*2^18 + c*2^10 + p
      int p = o & (HW - 1);
      int c = (o >> 10) & (EMB_DIM - 1);
      int bi = o >> 18;
      int idx = indices[bi * HW + p];
      float q  = emb[idx * EMB_DIM + c];      // L2-resident gather
      float zv = z[o];
      out[o] = q;                             // straight-through fwd value
      float d = q - zv;
      s += d * d;
    }
  }
#pragma unroll
  for (int m = 16; m >= 1; m >>= 1) s += __shfl_xor(s, m, 32);
  if ((threadIdx.x & 31) == 0) atomicAdd(lossAcc, (double)s);
}

// ------- Kernel D: finalize loss scalar ------------------------------------
__global__ void vq_loss_kernel(const double* __restrict__ lossAcc,
                               float* __restrict__ out, int nElem) {
  if (threadIdx.x == 0 && blockIdx.x == 0) {
    // q_latent + 0.25 * e_latent; both equal mean((q-z)^2) in fwd value.
    out[nElem] = (float)(1.25 * (*lossAcc / (double)nElem));
  }
}

// ---------------------------------------------------------------------------
extern "C" void kernel_launch(void* const* d_in, const int* in_sizes, int n_in,
                              void* d_out, int out_size, void* d_ws, size_t ws_size,
                              hipStream_t stream) {
  const float* z   = (const float*)d_in[0];   // [32,256,32,32]
  const float* emb = (const float*)d_in[1];   // [1024,256]
  float* out = (float*)d_out;                 // 8388608 quantized + 1 loss
  const int nElem = in_sizes[0];              // 8,388,608
  const int nRows = nElem / EMB_DIM;          // 32,768

  // workspace: embHi[1024*128]u32 | embLo[1024*128]u32 | enorm[1024]f32 |
  //            indices[32768]i32 | lossAcc f64   (~1.16 MB, all 16B aligned)
  char* wsb = (char*)d_ws;
  unsigned int* embHi = (unsigned int*)wsb;
  unsigned int* embLo = (unsigned int*)(wsb + (size_t)NUM_EMB * CPAIRS * 4);
  float* enorm        = (float*)(wsb + (size_t)NUM_EMB * CPAIRS * 8);
  int* indices        = (int*)(wsb + (size_t)NUM_EMB * CPAIRS * 8 + NUM_EMB * 4);
  double* lossAcc     = (double*)(wsb + (size_t)NUM_EMB * CPAIRS * 8 + NUM_EMB * 4
                                  + (size_t)nRows * 4);

  vq_prep_kernel<<<NUM_EMB, 128, 0, stream>>>(emb, embHi, embLo, enorm, lossAcc);

  const size_t ldsBytes = (size_t)2 * ROWS_PER_BLOCK * LDS_STRIDE * 4;
  vq_argmin_kernel<<<nRows / ROWS_PER_BLOCK, 256, ldsBytes, stream>>>(
      z, embHi, embLo, enorm, indices);

  vq_gather_kernel<<<nElem / (256 * 4), 256, 0, stream>>>(
      z, emb, indices, out, lossAcc, nElem);

  vq_loss_kernel<<<1, 1, 0, stream>>>(lossAcc, out, nElem);
}
